// GATEncoder_15977278341201
// MI455X (gfx1250) — matrix operand order repaired and checked by reading, not compiled, not run on hardware
//
#include <hip/hip_runtime.h>

#define IN_DIM 128
#define HID 128
#define OUTD 128
#define HEADS 4
#define NC1 (HEADS * HID)   // 512
#define NEG_SLOPE 0.2f
#define EPSA 1e-16f

typedef __attribute__((ext_vector_type(16))) __bf16 bf16x16;
typedef __attribute__((ext_vector_type(8)))  float  f32x8;
typedef __attribute__((ext_vector_type(4)))  unsigned u32x4;

// ---------- helpers ----------
__device__ __forceinline__ unsigned short f2bf(float f) {
    unsigned u = __float_as_uint(f);
    u += 0x7FFFu + ((u >> 16) & 1u);       // round-to-nearest-even
    return (unsigned short)(u >> 16);
}

// order-preserving float <-> uint encoding for atomic max
__device__ __forceinline__ unsigned encf(float f) {
    unsigned u = __float_as_uint(f);
    return (u & 0x80000000u) ? ~u : (u | 0x80000000u);
}
__device__ __forceinline__ float decf(unsigned e) {
    unsigned u = (e & 0x80000000u) ? (e & 0x7FFFFFFFu) : ~e;
    return __uint_as_float(u);
}

__device__ __forceinline__ bf16x16 load_frag(const unsigned short* p0,
                                             const unsigned short* p1) {
    union { u32x4 q[2]; bf16x16 v; } u;
    u.q[0] = *(const u32x4*)p0;
    u.q[1] = *(const u32x4*)p1;
    return u.v;
}

__device__ __forceinline__ float lrelu(float v) {
    return (v > 0.f) ? v : NEG_SLOPE * v;
}
__device__ __forceinline__ float elu(float v) {
    return (v > 0.f) ? v : (expf(v) - 1.f);
}

// ---------- fill / convert ----------
__global__ void fill_f32_kernel(float* p, float v, size_t n) {
    size_t i = (size_t)blockIdx.x * blockDim.x + threadIdx.x;
    if (i < n) p[i] = v;
}
__global__ void fill_u32_kernel(unsigned* p, unsigned v, size_t n) {
    size_t i = (size_t)blockIdx.x * blockDim.x + threadIdx.x;
    if (i < n) p[i] = v;
}
__global__ void cvt_bf16_kernel(const float* __restrict__ in,
                                unsigned short* __restrict__ out, size_t n) {
    size_t i = (size_t)blockIdx.x * blockDim.x + threadIdx.x;
    if (i < n) out[i] = f2bf(in[i]);
}
// W [K x NC] fp32 row-major -> Wt [NC x K] bf16
__global__ void cvt_tr_kernel(const float* __restrict__ W,
                              unsigned short* __restrict__ Wt, int K, int NC) {
    int i = blockIdx.x * blockDim.x + threadIdx.x;
    if (i >= K * NC) return;
    int k = i / NC, n = i % NC;
    Wt[(size_t)n * K + k] = f2bf(W[i]);
}

// ---------- WMMA GEMM (bf16 in, fp32 out) + fused attention dot products ----------
// A [M x K] bf16 row-major, Bt [NC x K] bf16 (B transposed).
// Each wave computes one 16x16 C tile; fused epilogue accumulates
// aS[m,h] += sum_d C[m, h*headDim+d]*attS[h*headDim+d] via wave32 shuffles.
__global__ __launch_bounds__(256) void wmma_gemm_att(
    const unsigned short* __restrict__ A,
    const unsigned short* __restrict__ Bt,
    float* __restrict__ C,
    const float* __restrict__ attS,  // [NC] flattened [heads, headDim]
    const float* __restrict__ attD,
    float* __restrict__ aS,          // [M x heads], must be pre-zeroed
    float* __restrict__ aD,
    int M, int K, int NC, int heads, int headDim)
{
    const int tilesN = NC >> 4;
    const int tilesM = (M + 15) >> 4;
    int wave = blockIdx.x * (blockDim.x >> 5) + (threadIdx.x >> 5);
    if (wave >= tilesM * tilesN) return;   // wave-uniform exit: EXEC stays all-1s
    int tm = wave / tilesN, tn = wave % tilesN;
    int m0 = tm << 4, n0 = tn << 4;

    int lane = threadIdx.x & 31;
    int half = lane >> 4;     // lane half selects K sub-block per ISA layout
    int l15  = lane & 15;

    int arow = m0 + l15; if (arow >= M) arow = M - 1;
    const unsigned short* arp = A  + (size_t)arow * K;
    const unsigned short* brp = Bt + (size_t)(n0 + l15) * K;

    f32x8 c = {};
    for (int kt = 0; kt < K; kt += 32) {
        // A 16x32: lane half 0 -> K {0..7,16..23}, half 1 -> K {8..15,24..31}
        bf16x16 a = load_frag(arp + kt + (half << 3),
                              arp + kt + (half << 3) + 16);
        // B 32x16 (from Bt): col = l15, lane half 0 -> K {0..7,16..23}, half 1 -> K {8..15,24..31}
        bf16x16 b = load_frag(brp + kt + (half << 3),
                              brp + kt + (half << 3) + 16);
        c = __builtin_amdgcn_wmma_f32_16x16x32_bf16(false, a, false, b,
                                                    (short)0, c, false, false);
    }

    // epilogue: store C tile + fused per-node attention coefficients
    float as = attS[n0 + l15];
    float ad = attD[n0 + l15];
    int head = n0 / headDim;   // headDim multiple of 16 -> tile within one head
    #pragma unroll
    for (int r = 0; r < 8; ++r) {
        int mrow = m0 + r + (half << 3);   // C layout: VGPR r, half -> M
        float cv = c[r];
        if (mrow < M) C[(size_t)mrow * NC + n0 + l15] = cv;
        float ps = cv * as;
        float pd = cv * ad;
        #pragma unroll
        for (int off = 1; off < 16; off <<= 1) {   // reduce within 16-lane group
            ps += __shfl_xor(ps, off, 32);
            pd += __shfl_xor(pd, off, 32);
        }
        if (l15 == 0 && mrow < M) {
            atomicAdd(&aS[(size_t)mrow * heads + head], ps);
            atomicAdd(&aD[(size_t)mrow * heads + head], pd);
        }
    }
}

// ---------- edge passes (self loops appended: ei >= E -> src=dst=ei-E) ----------
__global__ void edge_max_kernel(const int* __restrict__ src, const int* __restrict__ dst,
                                int E, int N, int heads,
                                const float* __restrict__ aS, const float* __restrict__ aD,
                                unsigned* __restrict__ mx)
{
    long long gid = (long long)blockIdx.x * blockDim.x + threadIdx.x;
    long long tot = (long long)(E + N) * heads;
    if (gid >= tot) return;
    int h = (int)(gid % heads);
    long long ei = gid / heads;
    int s, d;
    if (ei < E) { s = src[ei]; d = dst[ei]; } else { s = d = (int)(ei - E); }
    float v = lrelu(aS[(size_t)s * heads + h] + aD[(size_t)d * heads + h]);
    atomicMax(&mx[(size_t)d * heads + h], encf(v));
}

__global__ void edge_expsum_kernel(const int* __restrict__ src, const int* __restrict__ dst,
                                   int E, int N, int heads,
                                   const float* __restrict__ aS, const float* __restrict__ aD,
                                   const unsigned* __restrict__ mx, float* __restrict__ dn)
{
    long long gid = (long long)blockIdx.x * blockDim.x + threadIdx.x;
    long long tot = (long long)(E + N) * heads;
    if (gid >= tot) return;
    int h = (int)(gid % heads);
    long long ei = gid / heads;
    int s, d;
    if (ei < E) { s = src[ei]; d = dst[ei]; } else { s = d = (int)(ei - E); }
    size_t ih = (size_t)d * heads + h;
    float v = lrelu(aS[(size_t)s * heads + h] + aD[ih]);
    atomicAdd(&dn[ih], expf(v - decf(mx[ih])));
}

// one thread = (edge, head, 4 channels): float4 gather + 4 atomic f32 scatters
__global__ void edge_msg_kernel(const int* __restrict__ src, const int* __restrict__ dst,
                                int E, int N, int heads, int headDim,
                                const float* __restrict__ aS, const float* __restrict__ aD,
                                const unsigned* __restrict__ mx, const float* __restrict__ dn,
                                const float* __restrict__ Hm, float* __restrict__ acc)
{
    long long gid = (long long)blockIdx.x * blockDim.x + threadIdx.x;
    int lanes = headDim >> 2;
    long long tot = (long long)(E + N) * heads * lanes;
    if (gid >= tot) return;
    int c4 = (int)(gid % lanes);
    long long t = gid / lanes;
    int h = (int)(t % heads);
    long long ei = t / heads;
    int s, d;
    if (ei < E) { s = src[ei]; d = dst[ei]; } else { s = d = (int)(ei - E); }
    size_t ih = (size_t)d * heads + h;
    float v = lrelu(aS[(size_t)s * heads + h] + aD[ih]);
    float alpha = expf(v - decf(mx[ih])) / (dn[ih] + EPSA);
    const float4 hv = *(const float4*)(Hm + ((size_t)s * heads + h) * headDim + (c4 << 2));
    float* o = acc + ((size_t)d * heads + h) * headDim + (c4 << 2);
    atomicAdd(o + 0, hv.x * alpha);
    atomicAdd(o + 1, hv.y * alpha);
    atomicAdd(o + 2, hv.z * alpha);
    atomicAdd(o + 3, hv.w * alpha);
}

// ---------- epilogues ----------
__global__ void elu_bf16_kernel(const float* __restrict__ acc, const float* __restrict__ bias,
                                unsigned short* __restrict__ out, size_t n, int cols)
{
    size_t i = (size_t)blockIdx.x * blockDim.x + threadIdx.x;
    if (i >= n) return;
    out[i] = f2bf(elu(acc[i] + bias[i % cols]));
}

__global__ __launch_bounds__(128) void finalize_out_kernel(
    const float* __restrict__ acc, const float* __restrict__ bias,
    float* __restrict__ out, int N)
{
    __shared__ float red[4];
    int n = blockIdx.x;
    int t = threadIdx.x;
    float z = elu(acc[(size_t)n * OUTD + t] + bias[t]);
    float s = z * z;
    #pragma unroll
    for (int off = 16; off > 0; off >>= 1) s += __shfl_xor(s, off, 32);
    if ((t & 31) == 0) red[t >> 5] = s;
    __syncthreads();
    float nrm = fmaxf(sqrtf(red[0] + red[1] + red[2] + red[3]), 1e-12f);
    out[(size_t)n * OUTD + t] = z / nrm;
}

// ---------- host ----------
static inline unsigned cdiv_u(long long a, long long b) { return (unsigned)((a + b - 1) / b); }

extern "C" void kernel_launch(void* const* d_in, const int* in_sizes, int n_in,
                              void* d_out, int out_size, void* d_ws, size_t ws_size,
                              hipStream_t stream)
{
    const float* x     = (const float*)d_in[0];
    const int*   eidx  = (const int*)d_in[1];
    const float* W1    = (const float*)d_in[2];
    const float* attS1 = (const float*)d_in[3];
    const float* attD1 = (const float*)d_in[4];
    const float* b1    = (const float*)d_in[5];
    const float* W2    = (const float*)d_in[6];
    const float* attS2 = (const float*)d_in[7];
    const float* attD2 = (const float*)d_in[8];
    const float* b2    = (const float*)d_in[9];

    const int N = in_sizes[0] / IN_DIM;
    const int E = in_sizes[1] / 2;
    const int* srcI = eidx;
    const int* dstI = eidx + E;

    char* ws = (char*)d_ws;
    size_t o = 0;
    auto al = [](size_t v) { return (v + 255) & ~(size_t)255; };
    unsigned short* A1  = (unsigned short*)(ws + o); o += al((size_t)N * IN_DIM * 2);
    unsigned short* W1t = (unsigned short*)(ws + o); o += al((size_t)IN_DIM * NC1 * 2);
    unsigned short* W2t = (unsigned short*)(ws + o); o += al((size_t)NC1 * OUTD * 2);
    float*    Hbuf = (float*)(ws + o);          o += al((size_t)N * NC1 * 4);   // h1, reused for h2
    float*    ACC  = (float*)(ws + o);          o += al((size_t)N * NC1 * 4);   // scatter accumulators
    unsigned short* A2 = (unsigned short*)(ws + o); o += al((size_t)N * NC1 * 2);
    float*    aS = (float*)(ws + o);            o += al((size_t)N * HEADS * 4);
    float*    aD = (float*)(ws + o);            o += al((size_t)N * HEADS * 4);
    unsigned* mx = (unsigned*)(ws + o);         o += al((size_t)N * HEADS * 4);
    float*    dn = (float*)(ws + o);            o += al((size_t)N * HEADS * 4);

    const int TB = 256;
    const unsigned ENC_NEG_MAX = 0x00800000u;  // encf(-FLT_MAX)

    // conversions (fp32 -> bf16, weights transposed for contiguous B fragments)
    cvt_bf16_kernel<<<cdiv_u((long long)N * IN_DIM, TB), TB, 0, stream>>>(x, A1, (size_t)N * IN_DIM);
    cvt_tr_kernel<<<cdiv_u((long long)IN_DIM * NC1, TB), TB, 0, stream>>>(W1, W1t, IN_DIM, NC1);
    cvt_tr_kernel<<<cdiv_u((long long)NC1 * OUTD, TB), TB, 0, stream>>>(W2, W2t, NC1, OUTD);

    // ===== layer 1 =====
    size_t nh1 = (size_t)N * HEADS;
    fill_f32_kernel<<<cdiv_u(nh1, TB), TB, 0, stream>>>(aS, 0.f, nh1);
    fill_f32_kernel<<<cdiv_u(nh1, TB), TB, 0, stream>>>(aD, 0.f, nh1);
    fill_u32_kernel<<<cdiv_u(nh1, TB), TB, 0, stream>>>(mx, ENC_NEG_MAX, nh1);
    fill_f32_kernel<<<cdiv_u(nh1, TB), TB, 0, stream>>>(dn, 0.f, nh1);
    fill_f32_kernel<<<cdiv_u((long long)N * NC1, TB), TB, 0, stream>>>(ACC, 0.f, (size_t)N * NC1);

    {
        long long tiles = (long long)((N + 15) / 16) * (NC1 / 16);
        wmma_gemm_att<<<cdiv_u(tiles, 8), 256, 0, stream>>>(
            A1, W1t, Hbuf, attS1, attD1, aS, aD, N, IN_DIM, NC1, HEADS, HID);
    }
    long long totE1 = (long long)(E + N) * HEADS;
    edge_max_kernel<<<cdiv_u(totE1, TB), TB, 0, stream>>>(srcI, dstI, E, N, HEADS, aS, aD, mx);
    edge_expsum_kernel<<<cdiv_u(totE1, TB), TB, 0, stream>>>(srcI, dstI, E, N, HEADS, aS, aD, mx, dn);
    edge_msg_kernel<<<cdiv_u(totE1 * (HID / 4), TB), TB, 0, stream>>>(
        srcI, dstI, E, N, HEADS, HID, aS, aD, mx, dn, Hbuf, ACC);
    elu_bf16_kernel<<<cdiv_u((long long)N * NC1, TB), TB, 0, stream>>>(
        ACC, b1, A2, (size_t)N * NC1, NC1);

    // ===== layer 2 (heads=1) =====
    size_t nh2 = (size_t)N;
    fill_f32_kernel<<<cdiv_u(nh2, TB), TB, 0, stream>>>(aS, 0.f, nh2);
    fill_f32_kernel<<<cdiv_u(nh2, TB), TB, 0, stream>>>(aD, 0.f, nh2);
    fill_u32_kernel<<<cdiv_u(nh2, TB), TB, 0, stream>>>(mx, ENC_NEG_MAX, nh2);
    fill_f32_kernel<<<cdiv_u(nh2, TB), TB, 0, stream>>>(dn, 0.f, nh2);
    fill_f32_kernel<<<cdiv_u((long long)N * OUTD, TB), TB, 0, stream>>>(ACC, 0.f, (size_t)N * OUTD);

    {
        long long tiles = (long long)((N + 15) / 16) * (OUTD / 16);
        wmma_gemm_att<<<cdiv_u(tiles, 8), 256, 0, stream>>>(
            A2, W2t, Hbuf, attS2, attD2, aS, aD, N, NC1, OUTD, 1, OUTD);
    }
    long long totE2 = (long long)(E + N);
    edge_max_kernel<<<cdiv_u(totE2, TB), TB, 0, stream>>>(srcI, dstI, E, N, 1, aS, aD, mx);
    edge_expsum_kernel<<<cdiv_u(totE2, TB), TB, 0, stream>>>(srcI, dstI, E, N, 1, aS, aD, mx, dn);
    edge_msg_kernel<<<cdiv_u(totE2 * (OUTD / 4), TB), TB, 0, stream>>>(
        srcI, dstI, E, N, 1, OUTD, aS, aD, mx, dn, Hbuf, ACC);

    finalize_out_kernel<<<N, 128, 0, stream>>>(ACC, b2, (float*)d_out, N);
}
